// ImprovedTransformer_2164663517859
// MI455X (gfx1250) — compile-verified, matrix operand order checked by reading, not run on hardware
//
#include <hip/hip_runtime.h>
#include <cstdint>

typedef __bf16 bf16;
typedef __attribute__((ext_vector_type(16))) __bf16 v16bf;
typedef __attribute__((ext_vector_type(8)))  float  v8f;

#define D_MODEL 1024
#define NHEAD   16
#define HDIM    64
#define NLAYER  4
#define SEQ     2048
#define FFDIM   4096
#define VOCAB   32000
#define LN_EPS  1e-5f

// ---------------------------------------------------------------- helpers ---

__device__ __forceinline__ v8f wmma_bf16(v16bf a, v16bf b, v8f c) {
  // v_wmma_f32_16x16x32_bf16  (D = A x B + C), wave32
  return __builtin_amdgcn_wmma_f32_16x16x32_bf16(false, a, false, b,
                                                 (short)0, c, false, false);
}

// CDNA5 async global -> LDS copy (per-lane 16B), tracked with ASYNCcnt.
__device__ __forceinline__ void async_copy_b128(void* lds, const void* gptr) {
  uint32_t l = (uint32_t)(uintptr_t)lds;  // low 32 bits of LDS aperture addr = LDS offset
  asm volatile("global_load_async_to_lds_b128 %0, %1, off"
               :: "v"(l), "v"(gptr) : "memory");
}
__device__ __forceinline__ void wait_async0() {
  asm volatile("s_wait_asynccnt 0" ::: "memory");
}

// ---------------------------------------------------------------- embed -----

__global__ __launch_bounds__(256) void embed_kernel(
    const int* __restrict__ ids, const float* __restrict__ emb,
    float* __restrict__ x)
{
  long idx = (long)blockIdx.x * 256 + threadIdx.x;   // over S*D
  int s = (int)(idx >> 10);        // /D_MODEL
  int d = (int)(idx & 1023);
  x[idx] = emb[(long)ids[s] * D_MODEL + d];
}

// ---------------------------------------------------------------- layernorm -

__global__ __launch_bounds__(256) void ln_kernel(
    const float* __restrict__ x, const float* __restrict__ g,
    const float* __restrict__ b, bf16* __restrict__ out)
{
  __shared__ float red[256];
  __shared__ float red2[256];
  const int row = blockIdx.x;
  const int tid = threadIdx.x;
  const float* xr = x + (long)row * D_MODEL;

  float s = 0.f, s2 = 0.f;
  for (int j = tid; j < D_MODEL; j += 256) {
    float v = xr[j];
    s += v; s2 += v * v;
  }
  red[tid] = s; red2[tid] = s2;
  __syncthreads();
  for (int off = 128; off > 0; off >>= 1) {
    if (tid < off) { red[tid] += red[tid + off]; red2[tid] += red2[tid + off]; }
    __syncthreads();
  }
  const float mean = red[0] * (1.0f / D_MODEL);
  const float var  = red2[0] * (1.0f / D_MODEL) - mean * mean;
  const float rstd = rsqrtf(var + LN_EPS);
  for (int j = tid; j < D_MODEL; j += 256)
    out[(long)row * D_MODEL + j] = (bf16)((xr[j] - mean) * rstd * g[j] + b[j]);
}

// ---------------------------------------------------------------- gemm ------
// out[m,n] = act( A[m,:] . W[n,:] + bias[n] ) (+ resid[m,n])
// A : bf16 [M,K] row-major (activations), W : f32 [N,K] row-major (weights,
// converted to bf16 on the fly while staging into LDS).
// Block tile 128x128x32, 8 waves (4x2), wave tile 32x64 (2x4 WMMA tiles).
// LDS double-buffered; stage t+1 A-tile moved with async global->LDS DMA and
// stage t+1 W-tile kept in VGPRs across the WMMA block (software pipeline).
// Epilogue fully specialized at compile time (no runtime branches).

#define BM 128
#define BN 128
#define BK 32
#define LDK 40   // LDS row stride in bf16 elems (32 + 8 pad)

template<bool HAS_BIAS, bool GELU, bool HAS_RES, bool OUT_F, bool OUT_B>
__global__ __launch_bounds__(256) void gemm_bf16_kernel(
    const bf16* __restrict__ A, const float* __restrict__ W,
    const float* __restrict__ bias, const float* __restrict__ resid,
    float* __restrict__ outF, bf16* __restrict__ outB,
    int M, int N, int K)
{
  __shared__ bf16 As[2][BM][LDK];
  __shared__ bf16 Ws[2][BN][LDK];

  const int tid  = threadIdx.x;
  const int lane = tid & 31;
  const int wid  = tid >> 5;
  const int wm   = wid >> 1;        // 0..3  -> 32-row slab
  const int wn   = wid & 1;         // 0..1  -> 64-col slab
  const long bm  = (long)blockIdx.x * BM;
  const long bn  = (long)blockIdx.y * BN;

  const int lrow = tid >> 1;            // 0..127
  const int lcol = (tid & 1) << 4;      // 0 or 16

  v8f acc[2][4];
#pragma unroll
  for (int i = 0; i < 2; ++i)
#pragma unroll
    for (int j = 0; j < 4; ++j)
      acc[i][j] = (v8f){0.f,0.f,0.f,0.f,0.f,0.f,0.f,0.f};

  const bf16*  gA = A + (bm + lrow) * (long)K + lcol;
  const float* gW = W + (bn + lrow) * (long)K + lcol;

  // ---- prologue: stage 0 into buffer 0
  async_copy_b128(&As[0][lrow][lcol],     gA);
  async_copy_b128(&As[0][lrow][lcol + 8], gA + 8);
  {
    float4 f0 = *(const float4*)(gW + 0);
    float4 f1 = *(const float4*)(gW + 4);
    float4 f2 = *(const float4*)(gW + 8);
    float4 f3 = *(const float4*)(gW + 12);
    bf16* dst = &Ws[0][lrow][lcol];
    dst[0]=(bf16)f0.x; dst[1]=(bf16)f0.y; dst[2]=(bf16)f0.z; dst[3]=(bf16)f0.w;
    dst[4]=(bf16)f1.x; dst[5]=(bf16)f1.y; dst[6]=(bf16)f1.z; dst[7]=(bf16)f1.w;
    dst[8]=(bf16)f2.x; dst[9]=(bf16)f2.y; dst[10]=(bf16)f2.z; dst[11]=(bf16)f2.w;
    dst[12]=(bf16)f3.x; dst[13]=(bf16)f3.y; dst[14]=(bf16)f3.z; dst[15]=(bf16)f3.w;
  }

  const int nk = K / BK;
  for (int t = 0; t < nk; ++t) {
    const int cur = t & 1;
    const int nxt = cur ^ 1;
    // own stage-t async copies complete (issued last iteration / prologue)
    wait_async0();
    // all waves: stage-t data visible, and everyone finished reading buf `nxt`
    __syncthreads();

    float4 f0, f1, f2, f3;
    const bool more = (t + 1 < nk);
    if (more) {
      const long ko = (long)(t + 1) * BK;
      // next A tile: async DMA straight into the other LDS buffer
      async_copy_b128(&As[nxt][lrow][lcol],     gA + ko);
      async_copy_b128(&As[nxt][lrow][lcol + 8], gA + ko + 8);
      // next W tile: keep in VGPRs across the WMMA block
      const float* p = gW + ko;
      f0 = *(const float4*)(p + 0);
      f1 = *(const float4*)(p + 4);
      f2 = *(const float4*)(p + 8);
      f3 = *(const float4*)(p + 12);
      if (t + 2 < nk)
        __builtin_prefetch(gW + (long)(t + 2) * BK, 0, 3);
    }

    // ---- compute stage t
    v16bf af[2];
#pragma unroll
    for (int mt = 0; mt < 2; ++mt) {
      const int r = wm * 32 + mt * 16 + (lane & 15);
      af[mt] = *(const v16bf*)&As[cur][r][(lane >> 4) << 4];
    }
#pragma unroll
    for (int nt = 0; nt < 4; ++nt) {
      const int c = wn * 64 + nt * 16 + (lane & 15);
      v16bf bfr = *(const v16bf*)&Ws[cur][c][(lane >> 4) << 4];
#pragma unroll
      for (int mt = 0; mt < 2; ++mt)
        acc[mt][nt] = wmma_bf16(af[mt], bfr, acc[mt][nt]);
    }

    // ---- drain stage t+1 W tile into the other buffer
    if (more) {
      bf16* dst = &Ws[nxt][lrow][lcol];
      dst[0]=(bf16)f0.x; dst[1]=(bf16)f0.y; dst[2]=(bf16)f0.z; dst[3]=(bf16)f0.w;
      dst[4]=(bf16)f1.x; dst[5]=(bf16)f1.y; dst[6]=(bf16)f1.z; dst[7]=(bf16)f1.w;
      dst[8]=(bf16)f2.x; dst[9]=(bf16)f2.y; dst[10]=(bf16)f2.z; dst[11]=(bf16)f2.w;
      dst[12]=(bf16)f3.x; dst[13]=(bf16)f3.y; dst[14]=(bf16)f3.z; dst[15]=(bf16)f3.w;
    }
  }

  // epilogue: C layout — VGPR i <-> row (i + 8*(lane>=16)), col = lane&15
#pragma unroll
  for (int mt = 0; mt < 2; ++mt) {
#pragma unroll
    for (int nt = 0; nt < 4; ++nt) {
#pragma unroll
      for (int i = 0; i < 8; ++i) {
        const long gr = bm + wm * 32 + mt * 16 + i + ((lane >> 4) << 3);
        const long gc = bn + wn * 64 + nt * 16 + (lane & 15);
        float v = acc[mt][nt][i];
        if (HAS_BIAS) v += bias[gc];
        if (GELU)     v = 0.5f * v * (1.0f + erff(v * 0.70710678118f));
        if (HAS_RES)  v += resid[gr * N + gc];
        if (OUT_F)    outF[gr * N + gc] = v;
        if (OUT_B)    outB[gr * N + gc] = (bf16)v;
      }
    }
  }
}

// ---------------------------------------------------------------- attention -
// Flash-style: block = 4 waves, 64 query rows per block, one head per block.y.
// Online softmax in f32; WMMA for Q.K^T and P.V; ALiBi + causal mask.
// K/V tiles are L2-resident (4 MB per layer), so single-buffered staging.

__global__ __launch_bounds__(128) void attn_kernel(
    const bf16* __restrict__ Qm, const bf16* __restrict__ Km,
    const bf16* __restrict__ Vm, const float* __restrict__ slopes,
    bf16* __restrict__ Om)
{
  __shared__ bf16 Qs[64][72];
  __shared__ bf16 Ks[64][72];
  __shared__ bf16 Vt[64][72];          // transposed: Vt[hd][kv]
  __shared__ bf16 Ps[4][16][72];       // per-wave P tile (16 x 64)

  const int tid  = threadIdx.x;
  const int lane = tid & 31;
  const int wid  = tid >> 5;           // 0..3
  const int h    = blockIdx.y;
  const int qb   = blockIdx.x * 64;
  const float slope = slopes[h];
  const float scale = 0.125f;          // 1/sqrt(64)

  // stage Q tile (64 x 64) via async copies
  {
    const int r = tid >> 1;
    const int c = (tid & 1) * 32;
    const bf16* g = Qm + (long)(qb + r) * D_MODEL + h * HDIM + c;
    async_copy_b128(&Qs[r][c],      g);
    async_copy_b128(&Qs[r][c + 8],  g + 8);
    async_copy_b128(&Qs[r][c + 16], g + 16);
    async_copy_b128(&Qs[r][c + 24], g + 24);
  }
  wait_async0();
  __syncthreads();

  v16bf qf[2];
#pragma unroll
  for (int ks = 0; ks < 2; ++ks)
    qf[ks] = *(const v16bf*)&Qs[wid * 16 + (lane & 15)][ks * 32 + ((lane >> 4) << 4)];

  v8f o[4];
#pragma unroll
  for (int nt = 0; nt < 4; ++nt)
    o[nt] = (v8f){0.f,0.f,0.f,0.f,0.f,0.f,0.f,0.f};
  float rowmax[8], rowsum[8];
#pragma unroll
  for (int i = 0; i < 8; ++i) { rowmax[i] = -1e30f; rowsum[i] = 0.f; }

  const int ntiles = blockIdx.x + 1;   // causal: kv tiles 0..q-tile
  for (int t = 0; t < ntiles; ++t) {
    const int kb = t * 64;
    __syncthreads();
    {
      // K tile: async DMA
      const int r = tid >> 1;
      const int c = (tid & 1) * 32;
      const bf16* gk = Km + (long)(kb + r) * D_MODEL + h * HDIM + c;
      async_copy_b128(&Ks[r][c],      gk);
      async_copy_b128(&Ks[r][c + 8],  gk + 8);
      async_copy_b128(&Ks[r][c + 16], gk + 16);
      async_copy_b128(&Ks[r][c + 24], gk + 24);
      // V tile: each thread loads a kv-row pair, stores kv-packed dwords
      const int p  = tid >> 2;           // 0..31 -> kv rows 2p, 2p+1
      const int cv = (tid & 3) * 16;     // hd offset
      const bf16* gv0 = Vm + (long)(kb + 2 * p) * D_MODEL + h * HDIM + cv;
      const bf16* gv1 = gv0 + D_MODEL;
      union { uint4 u; bf16 e[8]; } a0, a1, b0, b1;
      a0.u = *(const uint4*)(gv0);
      a1.u = *(const uint4*)(gv0 + 8);
      b0.u = *(const uint4*)(gv1);
      b1.u = *(const uint4*)(gv1 + 8);
#pragma unroll
      for (int j = 0; j < 8; ++j) {
        union { uint32_t u; bf16 e[2]; } w0, w1;
        w0.e[0] = a0.e[j]; w0.e[1] = b0.e[j];
        w1.e[0] = a1.e[j]; w1.e[1] = b1.e[j];
        *(uint32_t*)&Vt[cv + j][2 * p]     = w0.u;
        *(uint32_t*)&Vt[cv + 8 + j][2 * p] = w1.u;
      }
    }
    wait_async0();
    __syncthreads();

    // scores S = Q K^T
    v8f sa[4];
#pragma unroll
    for (int nt = 0; nt < 4; ++nt)
      sa[nt] = (v8f){0.f,0.f,0.f,0.f,0.f,0.f,0.f,0.f};
#pragma unroll
    for (int ks = 0; ks < 2; ++ks) {
#pragma unroll
      for (int nt = 0; nt < 4; ++nt) {
        v16bf bfr = *(const v16bf*)&Ks[nt*16 + (lane&15)][ks*32 + ((lane>>4)<<4)];
        sa[nt] = wmma_bf16(qf[ks], bfr, sa[nt]);
      }
    }

    // ALiBi + causal mask, online softmax
    const int gi_base = qb + wid * 16 + ((lane >> 4) << 3);
    const int gj_base = kb + (lane & 15);
#pragma unroll
    for (int i = 0; i < 8; ++i) {
      const int gi = gi_base + i;
      float sv[4];
      float mloc = -1e30f;
#pragma unroll
      for (int nt = 0; nt < 4; ++nt) {
        const int gj = gj_base + nt * 16;
        float s = sa[nt][i] * scale - slope * (float)(gj - gi);
        if (gj > gi) s = -1e30f;
        sv[nt] = s;
        mloc = fmaxf(mloc, s);
      }
      mloc = fmaxf(mloc, __shfl_xor(mloc, 1));
      mloc = fmaxf(mloc, __shfl_xor(mloc, 2));
      mloc = fmaxf(mloc, __shfl_xor(mloc, 4));
      mloc = fmaxf(mloc, __shfl_xor(mloc, 8));
      const float mnew = fmaxf(rowmax[i], mloc);
      const float corr = __expf(rowmax[i] - mnew);
      rowmax[i] = mnew;
      float psum = 0.f;
#pragma unroll
      for (int nt = 0; nt < 4; ++nt) {
        const float p = __expf(sv[nt] - mnew);
        psum += p;
        Ps[wid][i + ((lane >> 4) << 3)][nt * 16 + (lane & 15)] = (bf16)p;
      }
      psum += __shfl_xor(psum, 1);
      psum += __shfl_xor(psum, 2);
      psum += __shfl_xor(psum, 4);
      psum += __shfl_xor(psum, 8);
      rowsum[i] = rowsum[i] * corr + psum;
#pragma unroll
      for (int nt = 0; nt < 4; ++nt) o[nt][i] *= corr;
    }

    // O += P V   (wave-local Ps; LDS ops of one wave are in order)
#pragma unroll
    for (int ks = 0; ks < 2; ++ks) {
      v16bf pf = *(const v16bf*)&Ps[wid][lane & 15][ks*32 + ((lane>>4)<<4)];
#pragma unroll
      for (int nt = 0; nt < 4; ++nt) {
        v16bf vf = *(const v16bf*)&Vt[nt*16 + (lane&15)][ks*32 + ((lane>>4)<<4)];
        o[nt] = wmma_bf16(pf, vf, o[nt]);
      }
    }
  }

  // normalize and write
#pragma unroll
  for (int nt = 0; nt < 4; ++nt) {
#pragma unroll
    for (int i = 0; i < 8; ++i) {
      const long gr = qb + wid * 16 + ((lane >> 4) << 3) + i;
      const int  gc = h * HDIM + nt * 16 + (lane & 15);
      Om[gr * D_MODEL + gc] = (bf16)(o[nt][i] / rowsum[i]);
    }
  }
}

// ---------------------------------------------------------------- driver ----

extern "C" void kernel_launch(void* const* d_in, const int* in_sizes, int n_in,
                              void* d_out, int out_size, void* d_ws, size_t ws_size,
                              hipStream_t stream)
{
  (void)in_sizes; (void)n_in; (void)out_size; (void)ws_size;
  const int*   ids    = (const int*)  d_in[0];
  const float* emb    = (const float*)d_in[1];
  const float* slopes = (const float*)d_in[2];
  const float* Wq     = (const float*)d_in[3];
  const float* Wk     = (const float*)d_in[4];
  const float* Wv     = (const float*)d_in[5];
  const float* Wo     = (const float*)d_in[6];
  const float* W1     = (const float*)d_in[7];
  const float* b1     = (const float*)d_in[8];
  const float* W2     = (const float*)d_in[9];
  const float* b2     = (const float*)d_in[10];
  const float* g1     = (const float*)d_in[11];
  const float* be1    = (const float*)d_in[12];
  const float* g2     = (const float*)d_in[13];
  const float* be2    = (const float*)d_in[14];
  const float* gf     = (const float*)d_in[15];
  const float* bff    = (const float*)d_in[16];

  char* ws = (char*)d_ws;
  float* x    = (float*)(ws);                          // S*D f32   : 8 MB
  bf16*  hbuf = (bf16*) (ws + (size_t)8  * 1024*1024); // S*D bf16  : 4 MB
  bf16*  qb   = (bf16*) (ws + (size_t)12 * 1024*1024);
  bf16*  kb   = (bf16*) (ws + (size_t)16 * 1024*1024);
  bf16*  vb   = (bf16*) (ws + (size_t)20 * 1024*1024);
  bf16*  ab   = (bf16*) (ws + (size_t)24 * 1024*1024);
  bf16*  ffb  = (bf16*) (ws + (size_t)28 * 1024*1024); // S*F bf16  : 16 MB

  embed_kernel<<<SEQ * D_MODEL / 256, 256, 0, stream>>>(ids, emb, x);

  const dim3 gProj(SEQ / BM, D_MODEL / BN);
  const dim3 gFF1 (SEQ / BM, FFDIM / BN);
  const dim3 gAttn(SEQ / 64, NHEAD);

  for (int l = 0; l < NLAYER; ++l) {
    const float* wq = Wq + (size_t)l * D_MODEL * D_MODEL;
    const float* wk = Wk + (size_t)l * D_MODEL * D_MODEL;
    const float* wv = Wv + (size_t)l * D_MODEL * D_MODEL;
    const float* wo = Wo + (size_t)l * D_MODEL * D_MODEL;
    const float* w1 = W1 + (size_t)l * FFDIM * D_MODEL;
    const float* w2 = W2 + (size_t)l * D_MODEL * FFDIM;

    ln_kernel<<<SEQ, 256, 0, stream>>>(x, g1 + l * D_MODEL, be1 + l * D_MODEL, hbuf);

    // q/k/v projections: bf16 out only
    gemm_bf16_kernel<false,false,false,false,true><<<gProj, 256, 0, stream>>>(
        hbuf, wq, nullptr, nullptr, nullptr, qb, SEQ, D_MODEL, D_MODEL);
    gemm_bf16_kernel<false,false,false,false,true><<<gProj, 256, 0, stream>>>(
        hbuf, wk, nullptr, nullptr, nullptr, kb, SEQ, D_MODEL, D_MODEL);
    gemm_bf16_kernel<false,false,false,false,true><<<gProj, 256, 0, stream>>>(
        hbuf, wv, nullptr, nullptr, nullptr, vb, SEQ, D_MODEL, D_MODEL);

    attn_kernel<<<gAttn, 128, 0, stream>>>(qb, kb, vb, slopes, ab);

    // x = x + attn_out @ Wo^T : residual + f32 out
    gemm_bf16_kernel<false,false,true,true,false><<<gProj, 256, 0, stream>>>(
        ab, wo, nullptr, x, x, nullptr, SEQ, D_MODEL, D_MODEL);

    ln_kernel<<<SEQ, 256, 0, stream>>>(x, g2 + l * D_MODEL, be2 + l * D_MODEL, hbuf);

    // ff = gelu(h2 @ W1^T + b1) : bias + gelu + bf16 out
    gemm_bf16_kernel<true,true,false,false,true><<<gFF1, 256, 0, stream>>>(
        hbuf, w1, b1 + (size_t)l * FFDIM, nullptr, nullptr, ffb, SEQ, FFDIM, D_MODEL);
    // x = x + ff @ W2^T + b2 : bias + residual + f32 out
    gemm_bf16_kernel<true,false,true,true,false><<<gProj, 256, 0, stream>>>(
        ffb, w2, b2 + (size_t)l * D_MODEL, x, x, nullptr, SEQ, D_MODEL, FFDIM);
  }

  ln_kernel<<<SEQ, 256, 0, stream>>>(x, gf, bff, hbuf);

  // logits = xf @ emb^T  (tied head) : f32 out
  const dim3 gHead(SEQ / BM, VOCAB / BN);
  gemm_bf16_kernel<false,false,false,true,false><<<gHead, 256, 0, stream>>>(
      hbuf, emb, nullptr, nullptr, (float*)d_out, nullptr, SEQ, VOCAB, D_MODEL);
}
